// HCMGNNBasedMetaPathModel_12300786335769
// MI455X (gfx1250) — matrix operand order repaired and checked by reading, not compiled, not run on hardware
//
#include <hip/hip_runtime.h>
#include <math.h>

#define NB 30000
#define NT 2000
#define DIM 256
#define NLAYER 3
#define NEDGE 600000
#define LN_EPS 1e-5f

typedef __attribute__((ext_vector_type(16))) __bf16 v16bf;
typedef __attribute__((ext_vector_type(8)))  float  v8f;

// ---------------- fragment helpers ----------------
// A 16x32 bf16 layout: elems 0..7 -> K=kb..kb+7, elems 8..15 -> K=kb+16..kb+23,
// kb = kk + (lane>=16 ? 8 : 0). All loads contiguous float4 (global_load_b128).
static __device__ __forceinline__ v16bf frag_a_fast(const float* __restrict__ p, int kb, float s) {
  const float4 q0 = *reinterpret_cast<const float4*>(p + kb + 0);
  const float4 q1 = *reinterpret_cast<const float4*>(p + kb + 4);
  const float4 q2 = *reinterpret_cast<const float4*>(p + kb + 16);
  const float4 q3 = *reinterpret_cast<const float4*>(p + kb + 20);
  v16bf f;
  f[0]  = (__bf16)(q0.x * s); f[1]  = (__bf16)(q0.y * s);
  f[2]  = (__bf16)(q0.z * s); f[3]  = (__bf16)(q0.w * s);
  f[4]  = (__bf16)(q1.x * s); f[5]  = (__bf16)(q1.y * s);
  f[6]  = (__bf16)(q1.z * s); f[7]  = (__bf16)(q1.w * s);
  f[8]  = (__bf16)(q2.x * s); f[9]  = (__bf16)(q2.y * s);
  f[10] = (__bf16)(q2.z * s); f[11] = (__bf16)(q2.w * s);
  f[12] = (__bf16)(q3.x * s); f[13] = (__bf16)(q3.y * s);
  f[14] = (__bf16)(q3.z * s); f[15] = (__bf16)(q3.w * s);
  return f;
}

// B 32x16 bf16 ("nt": B row-major [N,K]): elems 0..7 -> K=kh..kh+7, elems 8..15 -> K=kh+16..kh+23,
// kh = kk + (lane>=16 ? 8 : 0).
static __device__ __forceinline__ v16bf frag_b_nt_fast(const float* __restrict__ p, int kh) {
  const float4 q0 = *reinterpret_cast<const float4*>(p + kh + 0);
  const float4 q1 = *reinterpret_cast<const float4*>(p + kh + 4);
  const float4 q2 = *reinterpret_cast<const float4*>(p + kh + 16);
  const float4 q3 = *reinterpret_cast<const float4*>(p + kh + 20);
  v16bf f;
  f[0]  = (__bf16)q0.x; f[1]  = (__bf16)q0.y; f[2]  = (__bf16)q0.z; f[3]  = (__bf16)q0.w;
  f[4]  = (__bf16)q1.x; f[5]  = (__bf16)q1.y; f[6]  = (__bf16)q1.z; f[7]  = (__bf16)q1.w;
  f[8]  = (__bf16)q2.x; f[9]  = (__bf16)q2.y; f[10] = (__bf16)q2.z; f[11] = (__bf16)q2.w;
  f[12] = (__bf16)q3.x; f[13] = (__bf16)q3.y; f[14] = (__bf16)q3.z; f[15] = (__bf16)q3.w;
  return f;
}

// Branch-free masked tail (clamped contiguous loads + v_cndmask selects).
static __device__ __forceinline__ v16bf frag_b_nt_tail(const float* __restrict__ p, int kh, int K) {
  v16bf f;
#pragma unroll
  for (int j = 0; j < 16; ++j) {
    const int k = kh + (j < 8 ? j : j + 8);
    const int kc = k < K ? k : K - 1;
    const float v = p[kc];
    f[j] = (__bf16)(k < K ? v : 0.0f);
  }
  return f;
}

// ---------------- Generic WMMA GEMM (B in [N,K] "nt" form only) ----------------
// C = alpha*(A'*B^T) + bias + add (+ C).  REQUIRES: K % 32 == 0, M,N % 16 == 0.
// 128 threads = 4 waves; each wave owns a 16x32 output (two 16x16 tiles sharing the A frag).
__global__ __launch_bounds__(128)
void gemm_bf16_wmma(const float* __restrict__ A, int lda,
                    const float* __restrict__ rowscale,
                    const float* __restrict__ B, int ldb,
                    const float* __restrict__ bias,
                    float* __restrict__ C, int ldc,
                    int M, int N, int K, int accumulate,
                    const float* __restrict__ alpha_ptr,
                    const float* __restrict__ add_ptr) {
  const int wave = threadIdx.x >> 5;
  const int lane = threadIdx.x & 31;
  const int tm = blockIdx.x;
  const int ntiles = N >> 4;
  const int tn0 = (blockIdx.y * 4 + wave) * 2;
  if (tn0 >= ntiles) return;                 // wave-uniform
  const bool has2 = (tn0 + 1) < ntiles;      // wave-uniform
  const int lcol = lane & 15;
  const int arow = tm * 16 + lcol;
  const int ksel = (lane >> 4) << 3;         // A half-select: +0 / +8
  const int khalf = (lane >> 4) << 3;        // B half-select: +0 / +8
  const float rs = rowscale ? rowscale[arow] : 1.0f;
  const float* pa = A + (long)arow * lda;
  const int col0 = tn0 * 16 + lcol;
  const int col1 = col0 + 16;
  const float* pb0 = B + (long)col0 * ldb;
  const float* pb1 = B + (long)col1 * ldb;
  v8f acc0 = {0.f, 0.f, 0.f, 0.f, 0.f, 0.f, 0.f, 0.f};
  v8f acc1 = {0.f, 0.f, 0.f, 0.f, 0.f, 0.f, 0.f, 0.f};
  if (has2) {
    for (int kk = 0; kk < K; kk += 32) {
      const v16bf af = frag_a_fast(pa, kk + ksel, rs);
      const v16bf b0 = frag_b_nt_fast(pb0, kk + khalf);
      const v16bf b1 = frag_b_nt_fast(pb1, kk + khalf);
      acc0 = __builtin_amdgcn_wmma_f32_16x16x32_bf16(false, af, false, b0,
                                                     (short)0, acc0, false, false);
      acc1 = __builtin_amdgcn_wmma_f32_16x16x32_bf16(false, af, false, b1,
                                                     (short)0, acc1, false, false);
    }
  } else {
    for (int kk = 0; kk < K; kk += 32) {
      const v16bf af = frag_a_fast(pa, kk + ksel, rs);
      const v16bf b0 = frag_b_nt_fast(pb0, kk + khalf);
      acc0 = __builtin_amdgcn_wmma_f32_16x16x32_bf16(false, af, false, b0,
                                                     (short)0, acc0, false, false);
    }
  }
  const float alpha = alpha_ptr ? alpha_ptr[0] : 1.0f;
  const float adds  = add_ptr ? add_ptr[0] : 0.0f;
  const int mhalf = (lane >> 4) << 3;
#pragma unroll
  for (int g = 0; g < 8; ++g) {
    const int m = tm * 16 + mhalf + g;
    const long i0 = (long)m * ldc + col0;
    float v = acc0[g] * alpha + adds;
    if (bias) v += bias[col0];
    if (accumulate) v += C[i0];
    C[i0] = v;
  }
  if (has2) {
#pragma unroll
    for (int g = 0; g < 8; ++g) {
      const int m = tm * 16 + mhalf + g;
      const long i1 = (long)m * ldc + col1;
      float v = acc1[g] * alpha + adds;
      if (bias) v += bias[col1];
      if (accumulate) v += C[i1];
      C[i1] = v;
    }
  }
}

// ---------------- Metapath WMMA GEMM: C = (w0*adj0 + w1*adj1) @ XtT^T ----------------
// XtT is xt transposed to [N=DIM, K=NT] so B fragments are contiguous b128 loads.
// K (=2000) not a multiple of 32: unguarded main loop + branch-free masked tail.
__global__ __launch_bounds__(128)
void gemm_meta_wmma(const float* __restrict__ adj, long adj_stride,
                    const float* __restrict__ w2,
                    const float* __restrict__ XtT,  // [N][K] row-major
                    float* __restrict__ C, int ldc,
                    int M, int N, int K) {
  const int wave = threadIdx.x >> 5;
  const int lane = threadIdx.x & 31;
  const int tm = blockIdx.x;
  const int ntiles = N >> 4;
  const int tn0 = (blockIdx.y * 4 + wave) * 2;
  if (tn0 >= ntiles) return;
  const bool has2 = (tn0 + 1) < ntiles;
  const int lcol = lane & 15;
  const int arow = tm * 16 + lcol;
  const int ksel = (lane >> 4) << 3;
  const int khalf = (lane >> 4) << 3;
  const int col0 = tn0 * 16 + lcol;
  const int col1 = col0 + 16;
  const float w0 = w2[0], w1 = w2[1];
  const float* p0 = adj + (long)arow * K;
  const float* p1 = p0 + adj_stride;
  const float* pb0 = XtT + (long)col0 * K;
  const float* pb1 = XtT + (long)col1 * K;
  v8f acc0 = {0.f, 0.f, 0.f, 0.f, 0.f, 0.f, 0.f, 0.f};
  v8f acc1 = {0.f, 0.f, 0.f, 0.f, 0.f, 0.f, 0.f, 0.f};
  const int kfull = K & ~31;
  for (int kk = 0; kk < kfull; kk += 32) {
    __builtin_prefetch(p0 + kk + 256, 0, 0);  // streaming 480MB adjacency
    __builtin_prefetch(p1 + kk + 256, 0, 0);
    const int kb = kk + ksel;
    const float4 a0 = *reinterpret_cast<const float4*>(p0 + kb + 0);
    const float4 a1 = *reinterpret_cast<const float4*>(p0 + kb + 4);
    const float4 a2 = *reinterpret_cast<const float4*>(p0 + kb + 16);
    const float4 a3 = *reinterpret_cast<const float4*>(p0 + kb + 20);
    const float4 c0 = *reinterpret_cast<const float4*>(p1 + kb + 0);
    const float4 c1 = *reinterpret_cast<const float4*>(p1 + kb + 4);
    const float4 c2 = *reinterpret_cast<const float4*>(p1 + kb + 16);
    const float4 c3 = *reinterpret_cast<const float4*>(p1 + kb + 20);
    v16bf af;
    af[0]  = (__bf16)(w0 * a0.x + w1 * c0.x); af[1]  = (__bf16)(w0 * a0.y + w1 * c0.y);
    af[2]  = (__bf16)(w0 * a0.z + w1 * c0.z); af[3]  = (__bf16)(w0 * a0.w + w1 * c0.w);
    af[4]  = (__bf16)(w0 * a1.x + w1 * c1.x); af[5]  = (__bf16)(w0 * a1.y + w1 * c1.y);
    af[6]  = (__bf16)(w0 * a1.z + w1 * c1.z); af[7]  = (__bf16)(w0 * a1.w + w1 * c1.w);
    af[8]  = (__bf16)(w0 * a2.x + w1 * c2.x); af[9]  = (__bf16)(w0 * a2.y + w1 * c2.y);
    af[10] = (__bf16)(w0 * a2.z + w1 * c2.z); af[11] = (__bf16)(w0 * a2.w + w1 * c2.w);
    af[12] = (__bf16)(w0 * a3.x + w1 * c3.x); af[13] = (__bf16)(w0 * a3.y + w1 * c3.y);
    af[14] = (__bf16)(w0 * a3.z + w1 * c3.z); af[15] = (__bf16)(w0 * a3.w + w1 * c3.w);
    const v16bf b0 = frag_b_nt_fast(pb0, kk + khalf);
    acc0 = __builtin_amdgcn_wmma_f32_16x16x32_bf16(false, af, false, b0,
                                                   (short)0, acc0, false, false);
    if (has2) {
      const v16bf b1 = frag_b_nt_fast(pb1, kk + khalf);
      acc1 = __builtin_amdgcn_wmma_f32_16x16x32_bf16(false, af, false, b1,
                                                     (short)0, acc1, false, false);
    }
  }
  if (kfull < K) {  // masked tail chunk
    const int kb = kfull + ksel;
    v16bf af;
#pragma unroll
    for (int j = 0; j < 8; ++j) {
      const int k0 = kb + j;
      const int k1 = kb + 16 + j;
      const int k0c = k0 < K ? k0 : K - 1;
      const int k1c = k1 < K ? k1 : K - 1;
      const float v0 = w0 * p0[k0c] + w1 * p1[k0c];
      const float v1 = w0 * p0[k1c] + w1 * p1[k1c];
      af[j]     = (__bf16)(k0 < K ? v0 : 0.0f);
      af[j + 8] = (__bf16)(k1 < K ? v1 : 0.0f);
    }
    const v16bf b0 = frag_b_nt_tail(pb0, kfull + khalf, K);
    acc0 = __builtin_amdgcn_wmma_f32_16x16x32_bf16(false, af, false, b0,
                                                   (short)0, acc0, false, false);
    if (has2) {
      const v16bf b1 = frag_b_nt_tail(pb1, kfull + khalf, K);
      acc1 = __builtin_amdgcn_wmma_f32_16x16x32_bf16(false, af, false, b1,
                                                     (short)0, acc1, false, false);
    }
  }
  const int mhalf = (lane >> 4) << 3;
#pragma unroll
  for (int g = 0; g < 8; ++g) {
    const int m = tm * 16 + mhalf + g;
    C[(long)m * ldc + col0] = acc0[g];
  }
  if (has2) {
#pragma unroll
    for (int g = 0; g < 8; ++g) {
      const int m = tm * 16 + mhalf + g;
      C[(long)m * ldc + col1] = acc1[g];
    }
  }
}

// ---------------- Transpose xt [NT x DIM] -> xtT [DIM x NT] (LDS tiled) ----------------
__global__ void transpose_k(const float* __restrict__ in, float* __restrict__ out,
                            int rows, int cols) {  // in: [rows][cols], out: [cols][rows]
  __shared__ float tile[16][17];
  const int tx = threadIdx.x, ty = threadIdx.y;
  const int r0 = blockIdx.x * 16, c0 = blockIdx.y * 16;
  tile[ty][tx] = in[(long)(r0 + ty) * cols + (c0 + tx)];
  __syncthreads();
  out[(long)(c0 + ty) * rows + (r0 + tx)] = tile[tx][ty];
}

// ---------------- Row-wise kernels ----------------
static __device__ __forceinline__ float block_sum(float v, float* sd) {
  const int t = threadIdx.x;
  sd[t] = v;
  __syncthreads();
  for (int s = blockDim.x >> 1; s > 0; s >>= 1) {
    if (t < s) sd[t] += sd[t + s];
    __syncthreads();
  }
  const float r = sd[0];
  __syncthreads();
  return r;
}

__global__ void sage_finish_k(const float* __restrict__ cpre,
                              const float* __restrict__ tbuf,
                              const float* __restrict__ gam,
                              const float* __restrict__ bet,
                              float* __restrict__ out) {
  __shared__ float sd[DIM];
  const int r = blockIdx.x, j = threadIdx.x;
  const long base = (long)r * DIM + j;
  const float c = cpre[base];
  const float ss = block_sum(c * c, sd);
  const float inv = 1.0f / fmaxf(sqrtf(ss), 1e-12f);
  const float y = tbuf[base] + c * inv;
  const float mean = block_sum(y, sd) * (1.0f / DIM);
  const float d = y - mean;
  const float var = block_sum(d * d, sd) * (1.0f / DIM);
  out[base] = d * rsqrtf(var + LN_EPS) * gam[j] + bet[j];
}

__global__ void ln_inplace_k(float* __restrict__ h,
                             const float* __restrict__ gam,
                             const float* __restrict__ bet,
                             int do_relu) {
  __shared__ float sd[DIM];
  const int r = blockIdx.x, j = threadIdx.x;
  const long base = (long)r * DIM + j;
  const float x = h[base];
  const float mean = block_sum(x, sd) * (1.0f / DIM);
  const float d = x - mean;
  const float var = block_sum(d * d, sd) * (1.0f / DIM);
  float v = d * rsqrtf(var + LN_EPS) * gam[j] + bet[j];
  if (do_relu) v = fmaxf(v, 0.0f);
  h[base] = v;
}

__global__ void l2rows128_k(float* __restrict__ p) {  // blockDim.x == 128
  __shared__ float sd[128];
  const int r = blockIdx.x, j = threadIdx.x;
  const long base = (long)r * 128 + j;
  const float x = p[base];
  const float ss = block_sum(x * x, sd);
  p[base] = x / fmaxf(sqrtf(ss), 1e-12f);
}

// ---------------- Edge scatter (SAGE mean numerator + counts) ----------------
__global__ void edge_scatter_k(const int* __restrict__ src,
                               const int* __restrict__ dst,
                               const float* __restrict__ xsrc,
                               float* __restrict__ agg,
                               float* __restrict__ cnt,
                               int E) {
  const long t = (long)blockIdx.x * blockDim.x + threadIdx.x;
  const long e = t >> 6;                 // 64 threads per edge
  if (e >= E) return;
  const int jj = (int)(t & 63) << 2;     // 4 floats per thread
  const int s = src[e], d = dst[e];
  const float4 v = *reinterpret_cast<const float4*>(xsrc + (long)s * DIM + jj);
  float* p = agg + (long)d * DIM + jj;
  __hip_atomic_fetch_add(p + 0, v.x, __ATOMIC_RELAXED, __HIP_MEMORY_SCOPE_AGENT);
  __hip_atomic_fetch_add(p + 1, v.y, __ATOMIC_RELAXED, __HIP_MEMORY_SCOPE_AGENT);
  __hip_atomic_fetch_add(p + 2, v.z, __ATOMIC_RELAXED, __HIP_MEMORY_SCOPE_AGENT);
  __hip_atomic_fetch_add(p + 3, v.w, __ATOMIC_RELAXED, __HIP_MEMORY_SCOPE_AGENT);
  if (jj == 0)
    __hip_atomic_fetch_add(cnt + d, 1.0f, __ATOMIC_RELAXED, __HIP_MEMORY_SCOPE_AGENT);
}

// ---------------- Small helpers ----------------
__global__ void fill0_k(float* __restrict__ p, long n) {
  const long i = (long)blockIdx.x * blockDim.x + threadIdx.x;
  if (i < n) p[i] = 0.0f;
}
__global__ void recip_max1_k(float* __restrict__ p, int n) {
  const int i = blockIdx.x * blockDim.x + threadIdx.x;
  if (i < n) p[i] = 1.0f / fmaxf(p[i], 1.0f);
}
__global__ void add_inplace_k(float* __restrict__ a, const float* __restrict__ b, long n) {
  const long i = (long)blockIdx.x * blockDim.x + threadIdx.x;
  if (i < n) a[i] += b[i];
}
__global__ void softmax2_k(const float* __restrict__ w, float* __restrict__ o) {
  if (threadIdx.x == 0 && blockIdx.x == 0) {
    const float a = w[0], b = w[1];
    const float m = fmaxf(a, b);
    const float ea = __expf(a - m), eb = __expf(b - m);
    const float s = ea + eb;
    o[0] = ea / s;
    o[1] = eb / s;
  }
}

// ---------------- Host orchestration ----------------
extern "C" void kernel_launch(void* const* d_in, const int* in_sizes, int n_in,
                              void* d_out, int out_size, void* d_ws, size_t ws_size,
                              hipStream_t stream) {
  (void)in_sizes; (void)n_in; (void)out_size; (void)ws_size;
  const float* emb_b = (const float*)d_in[0];
  const float* emb_t = (const float*)d_in[1];
  const float* Wt_b  = (const float*)d_in[2];
  const float* bt_b  = (const float*)d_in[3];
  const float* Wt_t  = (const float*)d_in[4];
  const float* bt_t  = (const float*)d_in[5];
  const float* Wl_b  = (const float*)d_in[6];
  const float* bl_b  = (const float*)d_in[7];
  const float* Wr_b  = (const float*)d_in[8];
  const float* Wl_t  = (const float*)d_in[9];
  const float* bl_t  = (const float*)d_in[10];
  const float* Wr_t  = (const float*)d_in[11];
  const float* lng_b = (const float*)d_in[12];
  const float* lnb_b = (const float*)d_in[13];
  const float* lng_t = (const float*)d_in[14];
  const float* lnb_t = (const float*)d_in[15];
  const float* mp_w  = (const float*)d_in[16];
  const float* mp_adj= (const float*)d_in[17];
  const float* mpW   = (const float*)d_in[18];
  const float* mpb   = (const float*)d_in[19];
  const float* mplng = (const float*)d_in[20];
  const float* mplnb = (const float*)d_in[21];
  const float* Wp1b  = (const float*)d_in[22];
  const float* bp1b  = (const float*)d_in[23];
  const float* plngb = (const float*)d_in[24];
  const float* plnbb = (const float*)d_in[25];
  const float* Wp2b  = (const float*)d_in[26];
  const float* bp2b  = (const float*)d_in[27];
  const float* Wp1t  = (const float*)d_in[28];
  const float* bp1t  = (const float*)d_in[29];
  const float* plngt = (const float*)d_in[30];
  const float* plnbt = (const float*)d_in[31];
  const float* Wp2t  = (const float*)d_in[32];
  const float* bp2t  = (const float*)d_in[33];
  const float* temperature = (const float*)d_in[34];
  const float* sim_bias    = (const float*)d_in[35];
  const int* src_bt = (const int*)d_in[36];
  const int* dst_bt = (const int*)d_in[37];
  const int* src_tb = (const int*)d_in[38];
  const int* dst_tb = (const int*)d_in[39];

  float* ws = (float*)d_ws;
  size_t off = 0;
  auto carve = [&](size_t n) { float* p = ws + off; off += n; return p; };
  float* tb    = carve((size_t)NB * DIM);   // also hb_pre
  float* xb    = carve((size_t)NB * DIM);
  float* agg_b = carve((size_t)NB * DIM);   // also metapath weighted
  float* cb    = carve((size_t)NB * DIM);   // also mp_out, then hb2 [NB x 128]
  float* tt    = carve((size_t)NT * DIM);   // also ht_pre
  float* xt    = carve((size_t)NT * DIM);
  float* agg_t = carve((size_t)NT * DIM);   // also xtT [DIM x NT]
  float* ct    = carve((size_t)NT * DIM);   // also ht2 [NT x 128]
  float* cnt_b = carve(30720);
  float* cnt_t = carve(2048);
  float* w2    = carve(8);

  auto gemm = [&](const float* A, int lda, const float* rsc,
                  const float* B, int ldb, const float* bias,
                  float* C, int ldc, int M, int N, int K, int acc,
                  const float* alphap, const float* addp) {
    dim3 g((M + 15) / 16, ((N >> 4) + 7) / 8);  // 4 waves x 2 tiles per block
    gemm_bf16_wmma<<<g, 128, 0, stream>>>(A, lda, rsc, B, ldb, bias,
                                          C, ldc, M, N, K, acc, alphap, addp);
  };
  auto zero = [&](float* p, long n) {
    fill0_k<<<(unsigned)((n + 255) / 256), 256, 0, stream>>>(p, n);
  };

  const float* xb_cur = emb_b;
  const float* xt_cur = emb_t;
  const long WQ = (long)DIM * DIM;

  for (int i = 0; i < NLAYER; ++i) {
    gemm(xb_cur, DIM, nullptr, Wt_b + i * WQ, DIM, bt_b + i * DIM, tb, DIM, NB, DIM, DIM, 0, nullptr, nullptr);
    gemm(xt_cur, DIM, nullptr, Wt_t + i * WQ, DIM, bt_t + i * DIM, tt, DIM, NT, DIM, DIM, 0, nullptr, nullptr);
    zero(agg_b, (long)NB * DIM); zero(agg_t, (long)NT * DIM);
    zero(cnt_b, NB);             zero(cnt_t, NT);
    {
      const unsigned blocks = (unsigned)(((long)NEDGE * 64 + 255) / 256);
      edge_scatter_k<<<blocks, 256, 0, stream>>>(src_tb, dst_tb, tt, agg_b, cnt_b, NEDGE);
      edge_scatter_k<<<blocks, 256, 0, stream>>>(src_bt, dst_bt, tb, agg_t, cnt_t, NEDGE);
    }
    recip_max1_k<<<(NB + 255) / 256, 256, 0, stream>>>(cnt_b, NB);
    recip_max1_k<<<(NT + 255) / 256, 256, 0, stream>>>(cnt_t, NT);
    gemm(agg_b, DIM, cnt_b, Wl_b + i * WQ, DIM, bl_b + i * DIM, cb, DIM, NB, DIM, DIM, 0, nullptr, nullptr);
    gemm(tb,    DIM, nullptr, Wr_b + i * WQ, DIM, nullptr,      cb, DIM, NB, DIM, DIM, 1, nullptr, nullptr);
    gemm(agg_t, DIM, cnt_t, Wl_t + i * WQ, DIM, bl_t + i * DIM, ct, DIM, NT, DIM, DIM, 0, nullptr, nullptr);
    gemm(tt,    DIM, nullptr, Wr_t + i * WQ, DIM, nullptr,      ct, DIM, NT, DIM, DIM, 1, nullptr, nullptr);
    sage_finish_k<<<NB, DIM, 0, stream>>>(cb, tb, lng_b + i * DIM, lnb_b + i * DIM, xb);
    sage_finish_k<<<NT, DIM, 0, stream>>>(ct, tt, lng_t + i * DIM, lnb_t + i * DIM, xt);
    xb_cur = xb;
    xt_cur = xt;
  }

  // Metapath: weighted = xb + sum_m softmax(mp_w)[m] * adj_m @ xt  (softmax sums to 1)
  softmax2_k<<<1, 32, 0, stream>>>(mp_w, w2);
  {
    dim3 tg(NT / 16, DIM / 16);
    transpose_k<<<tg, dim3(16, 16), 0, stream>>>(xt, agg_t, NT, DIM);  // xtT = agg_t
  }
  {
    dim3 g(NB / 16, ((DIM >> 4) + 7) / 8);
    gemm_meta_wmma<<<g, 128, 0, stream>>>(mp_adj, (long)NB * NT, w2, agg_t,
                                          agg_b, DIM, NB, DIM, NT);
  }
  add_inplace_k<<<(unsigned)(((long)NB * DIM + 255) / 256), 256, 0, stream>>>(agg_b, xb, (long)NB * DIM);
  gemm(agg_b, DIM, nullptr, mpW, DIM, mpb, cb, DIM, NB, DIM, DIM, 0, nullptr, nullptr);
  ln_inplace_k<<<NB, DIM, 0, stream>>>(cb, mplng, mplnb, 0);

  // hb = relu(LN([xb, mp_out] @ Wp1b^T + bp1b)) @ Wp2b^T + bp2b
  gemm(xb, DIM, nullptr, Wp1b,       2 * DIM, bp1b,    tb, DIM, NB, DIM, DIM, 0, nullptr, nullptr);
  gemm(cb, DIM, nullptr, Wp1b + DIM, 2 * DIM, nullptr, tb, DIM, NB, DIM, DIM, 1, nullptr, nullptr);
  ln_inplace_k<<<NB, DIM, 0, stream>>>(tb, plngb, plnbb, 1);
  gemm(tb, DIM, nullptr, Wp2b, DIM, bp2b, cb, DIM / 2, NB, DIM / 2, DIM, 0, nullptr, nullptr);

  // ht = relu(LN(xt @ Wp1t^T + bp1t)) @ Wp2t^T + bp2t
  gemm(xt, DIM, nullptr, Wp1t, DIM, bp1t, tt, DIM, NT, DIM, DIM, 0, nullptr, nullptr);
  ln_inplace_k<<<NT, DIM, 0, stream>>>(tt, plngt, plnbt, 1);
  gemm(tt, DIM, nullptr, Wp2t, DIM, bp2t, ct, DIM / 2, NT, DIM / 2, DIM, 0, nullptr, nullptr);

  // L2-normalize rows, then sim = temperature * hb_n @ ht_n^T + sim_bias
  l2rows128_k<<<NB, 128, 0, stream>>>(cb);
  l2rows128_k<<<NT, 128, 0, stream>>>(ct);
  gemm(cb, DIM / 2, nullptr, ct, DIM / 2, nullptr,
       (float*)d_out, NT, NB, NT, DIM / 2, 0, temperature, sim_bias);
}
